// AttentionSubsample_25984552141376
// MI455X (gfx1250) — matrix operand order repaired and checked by reading, not compiled, not run on hardware
//
#include <hip/hip_runtime.h>

// ---------------------------------------------------------------------------
// AttentionSubsample for MI455X (gfx1250, wave32).
// All GEMMs use V_WMMA_F32_16X16X32_BF16 (bf16 inputs, f32 accumulate):
// the bf16 WMMA path has 8x the K-throughput of the f32 16x16x4 WMMA, which
// moves this kernel from compute-bound (~100us) to near the 23.3 TB/s HBM
// roofline (~11us for ~250MB of traffic).  f32->bf16 uses the native convert
// (static_cast<__bf16>) so the backend can emit packed v_cvt ops instead of
// a 3-VALU bit-twiddle per element.
// ---------------------------------------------------------------------------

typedef __attribute__((ext_vector_type(16))) __bf16 v16bf;
typedef __attribute__((ext_vector_type(8)))  float  v8f;
typedef unsigned short u16;
typedef unsigned int   u32;

union BfVec { v16bf v; __bf16 e[16]; u16 s[16]; uint4 q[2]; };

__device__ __forceinline__ u16 f2bf(float f) {
  return __builtin_bit_cast(u16, static_cast<__bf16>(f));
}

__device__ __forceinline__ v8f wmma_bf16(v16bf a, v16bf b, v8f c) {
  // 8 args: (neg_a, A, neg_b, B, c_mod, C, reuse_a, reuse_b)
  return __builtin_amdgcn_wmma_f32_16x16x32_bf16(false, a, false, b, (short)0, c,
                                                 false, false);
}

// A operand (16x32 bf16) from f32 row-major source.
// Lane L holds row M = L&15; K chunks [base, base+8) and [base+16, base+24),
// base = (L>=16 ? 8 : 0), per the 16-bit A layout in CDNA5 ISA.
__device__ __forceinline__ v16bf a_from_f32(const float* row, int kbase, int lane) {
  int k0 = kbase + ((lane >> 4) << 3);
  float4 f0 = *(const float4*)(row + k0);
  float4 f1 = *(const float4*)(row + k0 + 4);
  float4 f2 = *(const float4*)(row + k0 + 16);
  float4 f3 = *(const float4*)(row + k0 + 20);
  BfVec r;
  r.e[0]  = (__bf16)f0.x; r.e[1]  = (__bf16)f0.y; r.e[2]  = (__bf16)f0.z; r.e[3]  = (__bf16)f0.w;
  r.e[4]  = (__bf16)f1.x; r.e[5]  = (__bf16)f1.y; r.e[6]  = (__bf16)f1.z; r.e[7]  = (__bf16)f1.w;
  r.e[8]  = (__bf16)f2.x; r.e[9]  = (__bf16)f2.y; r.e[10] = (__bf16)f2.z; r.e[11] = (__bf16)f2.w;
  r.e[12] = (__bf16)f3.x; r.e[13] = (__bf16)f3.y; r.e[14] = (__bf16)f3.z; r.e[15] = (__bf16)f3.w;
  return r.v;
}

// A operand from bf16 row-major source (two 16B loads).
__device__ __forceinline__ v16bf a_from_bf16(const u16* row, int kbase, int lane) {
  int k0 = kbase + ((lane >> 4) << 3);
  BfVec r;
  r.q[0] = *(const uint4*)(row + k0);
  r.q[1] = *(const uint4*)(row + k0 + 16);
  return r.v;
}

// B operand (32x16 bf16): lane L holds column N = L&15; K chunks
// [base, base+8) and [base+16, base+24), base = kbase + (L>=16 ? 8 : 0).
// `colrow` points to the K-major row of the matrix supplying column N.
__device__ __forceinline__ v16bf b_from_bf16(const u16* colrow, int kbase, int lane) {
  const u16* p = colrow + kbase + ((lane >> 4) << 3);
  BfVec r;
  r.q[0] = *(const uint4*)(p);
  r.q[1] = *(const uint4*)(p + 16);
  return r.v;
}

// ------------------------- constants ---------------------------------------
#define BATCH 512
#define NTOK  196      // 14*14 key/value tokens
#define NQ    49       // 7*7 query tokens
#define CIN   256
#define HEADS 8
#define KD    16
#define VD    32
#define KVDIM 384
#define QDIM  128
#define VATTN 256
#define ODIM  384
#define KP    224      // NTOK padded to multiple of 32 (7 K-steps)

// ------------------------- prep kernel -------------------------------------
__global__ void __launch_bounds__(256) prep_kernel(
    const float* __restrict__ W_kv, const float* __restrict__ g_kv,
    const float* __restrict__ b_kv, const float* __restrict__ m_kv,
    const float* __restrict__ v_kv,
    const float* __restrict__ W_q,  const float* __restrict__ g_q,
    const float* __restrict__ b_q,  const float* __restrict__ m_q,
    const float* __restrict__ v_q,
    const float* __restrict__ W_p,  const float* __restrict__ g_p,
    const float* __restrict__ b_p,  const float* __restrict__ m_p,
    const float* __restrict__ v_p,
    const float* __restrict__ ab,   const int* __restrict__ bidx,
    u16* __restrict__ WkvB, u16* __restrict__ WqB, u16* __restrict__ WpB,
    float* __restrict__ s_kv, float* __restrict__ t_kv,
    float* __restrict__ s_q,  float* __restrict__ t_q,
    float* __restrict__ s_p,  float* __restrict__ t_p,
    float* __restrict__ biasF, u16* __restrict__ k_ws, u16* __restrict__ vT_ws) {
  int tid = blockIdx.x * blockDim.x + threadIdx.x;
  int stride = gridDim.x * blockDim.x;
  // fold BN: y*s + t  with s = g*rsqrt(v+eps), t = b - m*s
  for (int i = tid; i < 896; i += stride) {
    if (i < 384)      { float s = g_kv[i] * rsqrtf(v_kv[i] + 1e-5f); s_kv[i] = s; t_kv[i] = b_kv[i] - m_kv[i] * s; }
    else if (i < 512) { int c = i - 384; float s = g_q[c] * rsqrtf(v_q[c] + 1e-5f); s_q[c] = s; t_q[c] = b_q[c] - m_q[c] * s; }
    else              { int c = i - 512; float s = g_p[c] * rsqrtf(v_p[c] + 1e-5f); s_p[c] = s; t_p[c] = b_p[c] - m_p[c] * s; }
  }
  // weights -> bf16
  for (int i = tid; i < 229376; i += stride) {
    if (i < 98304)       WkvB[i] = f2bf(W_kv[i]);
    else if (i < 131072) WqB[i - 98304] = f2bf(W_q[i - 98304]);
    else                 WpB[i - 131072] = f2bf(W_p[i - 131072]);
  }
  // full bias table [h][q][kp] with -1e30 in padded columns (softmax mask)
  for (int i = tid; i < HEADS * NQ * KP; i += stride) {
    int hh = i / (NQ * KP); int rem = i % (NQ * KP);
    int qr = rem / KP, kp = rem % KP;
    biasF[i] = (kp < NTOK) ? ab[hh * NTOK + bidx[qr * NTOK + kp]] : -1e30f;
  }
  // zero-pad k rows and vT columns beyond NTOK
  for (int i = tid; i < BATCH * HEADS * (KP - NTOK) * KD; i += stride) {
    int bh = i / ((KP - NTOK) * KD); int rem = i % ((KP - NTOK) * KD);
    int m = NTOK + rem / KD, d = rem % KD;
    k_ws[((size_t)bh * KP + m) * KD + d] = 0;
  }
  for (int i = tid; i < BATCH * HEADS * VD * (KP - NTOK); i += stride) {
    int bh = i / (VD * (KP - NTOK)); int rem = i % (VD * (KP - NTOK));
    int vd = rem / (KP - NTOK), kp = NTOK + rem % (KP - NTOK);
    vT_ws[((size_t)bh * VD + vd) * KP + kp] = 0;
  }
}

// ------------------ KV GEMM: X(196x256) @ Wkv^T -> k, vT --------------------
__global__ void __launch_bounds__(256) kv_gemm(
    const float* __restrict__ x, const u16* __restrict__ Wb,
    const float* __restrict__ s, const float* __restrict__ t,
    u16* __restrict__ k_ws, u16* __restrict__ vT_ws) {
  int b = blockIdx.x;
  int w = threadIdx.x >> 5, lane = threadIdx.x & 31;
  int u = blockIdx.y * 8 + w;
  if (u >= 78) return;                       // 13 M-tiles * 6 N-groups
  int mtile = u % 13, ng = u / 13;
  const float* xb = x + (size_t)b * NTOK * CIN;
  int arow = mtile * 16 + (lane & 15); if (arow > NTOK - 1) arow = NTOK - 1;
  const float* aptr = xb + (size_t)arow * CIN;
  v8f acc[4] = {};
  for (int ks = 0; ks < 8; ++ks) {
    int kbase = ks * 32;
    v16bf a = a_from_f32(aptr, kbase, lane);
#pragma unroll
    for (int j = 0; j < 4; ++j) {
      int ocol = (ng * 4 + j) * 16 + (lane & 15);
      v16bf bm = b_from_bf16(Wb + (size_t)ocol * CIN, kbase, lane);
      acc[j] = wmma_bf16(a, bm, acc[j]);
    }
  }
  int n = lane & 15, half = lane >> 4;
#pragma unroll
  for (int j = 0; j < 4; ++j) {
    int o = (ng * 4 + j) * 16 + n;
    float sv = s[o], tv = t[o];
    int h = o / (KD + VD), c = o % (KD + VD);
#pragma unroll
    for (int r = 0; r < 8; ++r) {
      int m = mtile * 16 + r + half * 8;
      if (m >= NTOK) continue;
      u16 bv = f2bf(acc[j][r] * sv + tv);
      if (c < KD) k_ws[(((size_t)b * HEADS + h) * KP + m) * KD + c] = bv;
      else        vT_ws[(((size_t)b * HEADS + h) * VD + (c - KD)) * KP + m] = bv;
    }
  }
}

// ------------------ Q GEMM: Xsub(49x256) @ Wq^T -> q (scale folded) ---------
__global__ void __launch_bounds__(256) q_gemm(
    const float* __restrict__ x, const u16* __restrict__ Wb,
    const float* __restrict__ s, const float* __restrict__ t,
    u16* __restrict__ q_ws) {
  int b = blockIdx.x;
  int w = threadIdx.x >> 5, lane = threadIdx.x & 31;
  int u = blockIdx.y * 8 + w;                // 32 tiles = 4 M x 8 N
  int mtile = u & 3, ntile = u >> 2;
  int tk = mtile * 16 + (lane & 15); if (tk > NQ - 1) tk = NQ - 1;
  int xrow = 28 * (tk / 7) + 2 * (tk % 7);   // stride-2 subsample of 14x14
  const float* aptr = x + ((size_t)b * NTOK + xrow) * CIN;
  int ocol = ntile * 16 + (lane & 15);
  v8f acc = {};
  for (int ks = 0; ks < 8; ++ks) {
    v16bf a  = a_from_f32(aptr, ks * 32, lane);
    v16bf bm = b_from_bf16(Wb + (size_t)ocol * CIN, ks * 32, lane);
    acc = wmma_bf16(a, bm, acc);
  }
  int n = lane & 15, half = lane >> 4;
  int o = ntile * 16 + n;
  float sv = s[o] * 0.25f, tv = t[o] * 0.25f;  // fold SCALE = KD^-0.5 = 1/4
  int h = o / KD, d = o % KD;
#pragma unroll
  for (int r = 0; r < 8; ++r) {
    int m = mtile * 16 + r + half * 8;         // rows 49..63 hold clamped dups
    q_ws[(((size_t)b * HEADS + h) * 64 + m) * KD + d] = f2bf(acc[r] * sv + tv);
  }
}

// ------------------ attention: one block per (b, h) -------------------------
// LDS buffer is dual-use: f32 logits [64][224] for softmax; each row's
// normalized probabilities are then re-packed IN PLACE as bf16 at the row's
// own base (u16 stride 448 per row), so stage 3's A-operand is two pure
// ds_load_b128 per K-step with no converts in the GEMM loop.
__global__ void __launch_bounds__(256) attn_kernel(
    const u16* __restrict__ q_ws, const u16* __restrict__ k_ws,
    const u16* __restrict__ vT_ws, const float* __restrict__ biasF,
    u16* __restrict__ o_ws) {
  __shared__ float probs[64 * KP];            // 57,344 B
  int b = blockIdx.x, h = blockIdx.y;
  int w = threadIdx.x >> 5, lane = threadIdx.x & 31;
  const u16* qb = q_ws + ((size_t)b * HEADS + h) * 64 * KD;
  const u16* kb = k_ws + ((size_t)b * HEADS + h) * KP * KD;
  const u16* vb = vT_ws + ((size_t)b * HEADS + h) * VD * KP;
  const float* bih = biasF + (size_t)h * NQ * KP;

  // Stage 1: logits = q @ k^T + bias.  K-dim is 16, zero-padded to 32.
  for (int uu = w; uu < 56; uu += 8) {        // 4 M-tiles x 14 N-tiles
    int mt = uu / 14, nt = uu % 14;
    BfVec av;
    {
      int qrow = mt * 16 + (lane & 15);
      av.q[0] = *(const uint4*)(qb + (size_t)qrow * KD + ((lane >> 4) << 3));
      av.q[1] = make_uint4(0, 0, 0, 0);       // K=16..31 zero
    }
    BfVec bv;
    {
      bv.q[0] = *(const uint4*)(kb + (size_t)(nt * 16 + (lane & 15)) * KD + ((lane >> 4) << 3));
      bv.q[1] = make_uint4(0, 0, 0, 0);       // K=16..31 zero
    }
    v8f c = {};
    c = wmma_bf16(av.v, bv.v, c);
    int n = lane & 15, half = lane >> 4;
#pragma unroll
    for (int r = 0; r < 8; ++r) {
      int qr = mt * 16 + r + half * 8;
      int kp = nt * 16 + n;
      float bias = (qr < NQ) ? bih[qr * KP + kp] : 0.0f;
      probs[qr * KP + kp] = c[r] + bias;
    }
  }
  __syncthreads();

  // Stage 2: softmax, one wave32 per row; 224 = 32 lanes x 7.
  // Each row is fully read into registers before its in-place bf16 rewrite.
  for (int r = w; r < NQ; r += 8) {
    float vals[7]; float vmax = -3.4e38f;
#pragma unroll
    for (int i = 0; i < 7; ++i) { vals[i] = probs[r * KP + lane + 32 * i]; vmax = fmaxf(vmax, vals[i]); }
#pragma unroll
    for (int off = 16; off >= 1; off >>= 1) vmax = fmaxf(vmax, __shfl_xor(vmax, off, 32));
    float ssum = 0.f;
#pragma unroll
    for (int i = 0; i < 7; ++i) { vals[i] = __expf(vals[i] - vmax); ssum += vals[i]; }
#pragma unroll
    for (int off = 16; off >= 1; off >>= 1) ssum += __shfl_xor(ssum, off, 32);
    float inv = 1.0f / ssum;
    u16* prow = (u16*)(probs + (size_t)r * KP);   // bf16 row at its own base
#pragma unroll
    for (int i = 0; i < 7; ++i) prow[lane + 32 * i] = f2bf(vals[i] * inv);
  }
  __syncthreads();

  // Stage 3: o = probs @ v.  K = 224 (7 steps); 8 tiles = 4 M x 2 N, 1/wave.
  {
    int mt = w >> 1, nt = w & 1;
    const u16* arow = (const u16*)probs + (size_t)(mt * 16 + (lane & 15)) * (2 * KP);
    const u16* brow = vb + (size_t)(nt * 16 + (lane & 15)) * KP;
    v8f c = {};
    for (int ks = 0; ks < 7; ++ks) {
      v16bf a  = a_from_bf16(arow, ks * 32, lane);
      v16bf bm = b_from_bf16(brow, ks * 32, lane);
      c = wmma_bf16(a, bm, c);
    }
    int n = lane & 15, half = lane >> 4;
#pragma unroll
    for (int r = 0; r < 8; ++r) {
      int qr = mt * 16 + r + half * 8;
      if (qr >= NQ) continue;
      float v = c[r];
      v = v * fminf(fmaxf(v + 3.0f, 0.0f), 6.0f) * (1.0f / 6.0f);  // hardswish
      o_ws[((size_t)b * NQ + qr) * VATTN + h * VD + nt * 16 + n] = f2bf(v);
    }
  }
}

// ------------------ proj GEMM: (25088x256) @ Wp^T -> out (f32) --------------
__global__ void __launch_bounds__(256) proj_gemm(
    const u16* __restrict__ o_ws, const u16* __restrict__ Wb,
    const float* __restrict__ s, const float* __restrict__ t,
    float* __restrict__ out) {
  int w = threadIdx.x >> 5, lane = threadIdx.x & 31;
  int u = blockIdx.x * 8 + w;                 // 1568 M-tiles x 6 N-groups
  int mtile = u / 6, ng = u % 6;
  const u16* aptr = o_ws + ((size_t)mtile * 16 + (lane & 15)) * VATTN;
  v8f acc[4] = {};
  for (int ks = 0; ks < 8; ++ks) {
    v16bf a = a_from_bf16(aptr, ks * 32, lane);
#pragma unroll
    for (int j = 0; j < 4; ++j) {
      int ocol = (ng * 4 + j) * 16 + (lane & 15);
      v16bf bm = b_from_bf16(Wb + (size_t)ocol * VATTN, ks * 32, lane);
      acc[j] = wmma_bf16(a, bm, acc[j]);
    }
  }
  int n = lane & 15, half = lane >> 4;
#pragma unroll
  for (int j = 0; j < 4; ++j) {
    int o = (ng * 4 + j) * 16 + n;
    float sv = s[o], tv = t[o];
#pragma unroll
    for (int r = 0; r < 8; ++r) {
      int m = mtile * 16 + r + half * 8;      // 25088 rows, exact multiple of 16
      out[(size_t)m * ODIM + o] = acc[j][r] * sv + tv;
    }
  }
}

// ---------------------------------------------------------------------------
extern "C" void kernel_launch(void* const* d_in, const int* in_sizes, int n_in,
                              void* d_out, int out_size, void* d_ws, size_t ws_size,
                              hipStream_t stream) {
  const float* x    = (const float*)d_in[0];
  const float* W_kv = (const float*)d_in[1];
  const float* g_kv = (const float*)d_in[2];
  const float* b_kv = (const float*)d_in[3];
  const float* m_kv = (const float*)d_in[4];
  const float* v_kv = (const float*)d_in[5];
  const float* W_q  = (const float*)d_in[6];
  const float* g_q  = (const float*)d_in[7];
  const float* b_q  = (const float*)d_in[8];
  const float* m_q  = (const float*)d_in[9];
  const float* v_q  = (const float*)d_in[10];
  const float* W_p  = (const float*)d_in[11];
  const float* g_p  = (const float*)d_in[12];
  const float* b_p  = (const float*)d_in[13];
  const float* m_p  = (const float*)d_in[14];
  const float* v_p  = (const float*)d_in[15];
  const float* ab   = (const float*)d_in[16];
  const int*   bidx = (const int*)d_in[17];

  char* ws = (char*)d_ws;
  size_t off = 0;
  auto alloc = [&](size_t bytes) -> char* {
    char* p = ws + off; off += (bytes + 255) & ~(size_t)255; return p;
  };
  u16*   WkvB  = (u16*)alloc((size_t)KVDIM * CIN * 2);
  u16*   WqB   = (u16*)alloc((size_t)QDIM * CIN * 2);
  u16*   WpB   = (u16*)alloc((size_t)ODIM * VATTN * 2);
  float* s_kv  = (float*)alloc(KVDIM * 4);
  float* t_kv  = (float*)alloc(KVDIM * 4);
  float* s_q   = (float*)alloc(QDIM * 4);
  float* t_q   = (float*)alloc(QDIM * 4);
  float* s_p   = (float*)alloc(ODIM * 4);
  float* t_p   = (float*)alloc(ODIM * 4);
  float* biasF = (float*)alloc((size_t)HEADS * NQ * KP * 4);
  u16*   k_ws  = (u16*)alloc((size_t)BATCH * HEADS * KP * KD * 2);
  u16*   vT_ws = (u16*)alloc((size_t)BATCH * HEADS * VD * KP * 2);
  u16*   q_ws  = (u16*)alloc((size_t)BATCH * HEADS * 64 * KD * 2);
  u16*   o_ws  = (u16*)alloc((size_t)BATCH * NQ * VATTN * 2);
  (void)ws_size; (void)in_sizes; (void)n_in; (void)out_size;

  prep_kernel<<<2048, 256, 0, stream>>>(
      W_kv, g_kv, b_kv, m_kv, v_kv, W_q, g_q, b_q, m_q, v_q,
      W_p, g_p, b_p, m_p, v_p, ab, bidx,
      WkvB, WqB, WpB, s_kv, t_kv, s_q, t_q, s_p, t_p, biasF, k_ws, vT_ws);
  kv_gemm<<<dim3(BATCH, 10), 256, 0, stream>>>(x, WkvB, s_kv, t_kv, k_ws, vT_ws);
  q_gemm<<<dim3(BATCH, 4), 256, 0, stream>>>(x, WqB, s_q, t_q, q_ws);
  attn_kernel<<<dim3(BATCH, HEADS), 256, 0, stream>>>(q_ws, k_ws, vT_ws, biasF, o_ws);
  proj_gemm<<<(BATCH * NQ / 16) * 6 / 8, 256, 0, stream>>>(o_ws, WpB, s_p, t_p,
                                                           (float*)d_out);
}
